// TIMNET_67061619360201
// MI455X (gfx1250) — compile-verified, matrix operand order checked by reading, not run on hardware
//
#include <hip/hip_runtime.h>

#define B_ 256
#define T_ 2048
#define C_ 39
#define CP 40          // padded channels per time-row in LDS ([t][c] layout)
#define NB_ 8
#define NC_ 8
#define THREADS 256
#define WAVES 8
#define PCH 48         // padded channel count for partial-sum rows

typedef __attribute__((ext_vector_type(8)))  float        v8f;
typedef __attribute__((ext_vector_type(16))) __bf16       v16bf;
typedef __attribute__((ext_vector_type(4)))  unsigned int v4u;

union BF16x16 { unsigned short u[16]; unsigned int w[8]; v16bf v; };

__device__ __forceinline__ unsigned short f2bf(float f) {
  unsigned int u = __builtin_bit_cast(unsigned int, f);
  u += 0x7FFFu + ((u >> 16) & 1u);          // round-to-nearest-even (setup only)
  return (unsigned short)(u >> 16);
}
__device__ __forceinline__ float bf2f(unsigned short h) {
  unsigned int u = ((unsigned int)h) << 16;
  return __builtin_bit_cast(float, u);
}
// pack two floats to packed bf16 (round-half-up) in one v_perm_b32
__device__ __forceinline__ unsigned int pack_bf16(float lo, float hi) {
  const unsigned int a = __builtin_bit_cast(unsigned int, lo) + 0x8000u;
  const unsigned int b = __builtin_bit_cast(unsigned int, hi) + 0x8000u;
  return __builtin_amdgcn_perm(b, a, 0x07060302u);  // {b[31:16], a[31:16]}
}

// Load one 8-channel run (16 bytes) from the [t][c] LDS tile; masked runs
// (causal t<0 or all-pad channel block) collapse to zero without branches.
__device__ __forceinline__ v4u ldrun(const unsigned short* src, int ci0, int ts, bool ok) {
  const int ci = ok ? ci0 : 0;
  const int tt = ok ? ts : 0;
  v4u val = *(const v4u*)(src + tt * CP + ci);
  const v4u z = {};
  return ok ? val : z;
}

// One conv stage over the full resident [T_][CP] tile.
// GEMM: M = Cout (3 tiles of 16), K = 48*tap + ci packed (3 chunks of 32), N = t.
// BN scale pre-folded into A; epilogue is add + activation only.
// MODE: 0 = conv1x1 (bias only, in-place X->X; tap-0 chunks skipped)
//       1 = conv + BN + ReLU            (X -> Y)
//       2 = conv + BN + sigmoid gating  (Y -> X, accumulate temporal sums)
template <int MODE>
__device__ void conv_stage(const unsigned short* src, unsigned short* dst,
                           const float* __restrict__ W,
                           const float* __restrict__ bias,
                           const float* __restrict__ gamma,
                           const float* __restrict__ beta,
                           const float* __restrict__ mean,
                           const float* __restrict__ var,
                           int d, float* __restrict__ partialRow) {
  const int tid  = threadIdx.x;
  const int lane = tid & 31;
  const int wave = tid >> 5;
  const int lm   = lane & 15;
  const int hi   = (lane >> 4) & 1;
  const int QS   = (MODE == 0) ? 1 : 0;   // first live K-chunk

  // ---- A fragments. Run j (ci0 = 16j+8hi) maps: tap0 -> chunk/slot
  //      {(0,0),(0,8),(1,0)}[j], tap1 -> {(1,8),(2,0),(2,8)}[j].
  //      One float2 load supplies both taps of one ci.
  BF16x16 A[3][3];
  const int q0s[3] = {0, 0, 1}, e0s[3] = {0, 8, 0};
  const int q1s[3] = {1, 2, 2}, e1s[3] = {8, 0, 8};
#pragma unroll
  for (int m = 0; m < 3; ++m) {
    const int co = 16 * m + lm;
    float sa = 1.f;
    if (MODE != 0) sa = (co < C_) ? gamma[co] * rsqrtf(var[co] + 1e-5f) : 0.f;
#pragma unroll
    for (int j = 0; j < 3; ++j) {
      const int ci0 = 16 * j + 8 * hi;
#pragma unroll
      for (int i = 0; i < 8; ++i) {
        const int ci = ci0 + i;
        float w0 = 0.f, w1 = 0.f;
        if (co < C_ && ci < C_) {
          if (MODE == 0) {
            w1 = W[co * C_ + ci];
          } else {
            const float2 wv = *(const float2*)(W + (size_t)(co * C_ + ci) * 2);
            w0 = wv.x * sa;
            w1 = wv.y * sa;
          }
        }
        A[m][q0s[j]].u[e0s[j] + i] = f2bf(w0);
        A[m][q1s[j]].u[e1s[j] + i] = f2bf(w1);
      }
    }
  }

  // ---- folded BN shift for this lane's 24 output channels (co = 16m+8hi+v)
  float sh[3][8], sm[3][8];
#pragma unroll
  for (int m = 0; m < 3; ++m)
#pragma unroll
    for (int v = 0; v < 8; ++v) {
      const int co = 16 * m + 8 * hi + v;
      sm[m][v] = 0.f;
      if (MODE == 0) {
        sh[m][v] = (co < C_) ? bias[co] : 0.f;
      } else if (co < C_) {
        const float s = gamma[co] * rsqrtf(var[co] + 1e-5f);
        sh[m][v] = beta[co] + s * (bias[co] - mean[co]);
      } else { sh[m][v] = 0.f; }
    }

  // ---- time-tile loop: each wave owns disjoint 16-wide column groups
#pragma unroll 2
  for (int tile = wave; tile < T_ / 16; tile += WAVES) {
    const int t   = tile * 16 + lm;
    const int tm  = t - d;            // tap-0 (delayed) source row
    const bool tk = (tm >= 0);
    const bool pk = (hi == 0);        // run ci0=32+8hi stays inside CP only for hi=0

    // B fragment: contiguous 8-channel runs (b128 each); tap-0 runs dead for MODE 0
    v4u r0 = {}, r1 = {}, r2 = {};
    if (MODE != 0) {
      r0 = ldrun(src,  8 * hi,     tm, tk);        // q0: K 8hi..
      r1 = ldrun(src, 16 + 8 * hi, tm, tk);        // q0: K 16+8hi..
      r2 = ldrun(src, 32 + 8 * hi, tm, tk && pk);  // q1: K 32+8hi..
    }
    const v4u r3 = ldrun(src,  8 * hi,     t, true);  // q1: K 48+8hi.. (tap1)
    const v4u r4 = ldrun(src, 16 + 8 * hi, t, true);  // q2: K 64+8hi..
    const v4u r5 = ldrun(src, 32 + 8 * hi, t, pk);    // q2: K 80+8hi..

    BF16x16 Bf[3];
    *(v4u*)&Bf[0].w[0] = r0;  *(v4u*)&Bf[0].w[4] = r1;
    *(v4u*)&Bf[1].w[0] = r2;  *(v4u*)&Bf[1].w[4] = r3;
    *(v4u*)&Bf[2].w[0] = r4;  *(v4u*)&Bf[2].w[4] = r5;

    v8f acc[3] = {};
#pragma unroll
    for (int m = 0; m < 3; ++m)
#pragma unroll
      for (int q = QS; q < 3; ++q)
        acc[m] = __builtin_amdgcn_wmma_f32_16x16x32_bf16(
            false, A[m][q].v, false, Bf[q].v, (short)0, acc[m], false, false);

    // ---- epilogue: lane owns 8 consecutive channels per M-tile at time t
#pragma unroll
    for (int m = 0; m < 3; ++m) {
      const int co0 = 16 * m + 8 * hi;
      if (co0 < CP) {
        unsigned short* p = dst + t * CP + co0;
        v4u xin = {};
        if (MODE == 2) xin = *(const v4u*)p;
        v4u o;
#pragma unroll
        for (int j = 0; j < 4; ++j) {
          float v0 = acc[m][2 * j]     + sh[m][2 * j];
          float v1 = acc[m][2 * j + 1] + sh[m][2 * j + 1];
          if (MODE == 1) { v0 = fmaxf(v0, 0.f); v1 = fmaxf(v1, 0.f); }
          if (MODE == 2) {
            const float x0 = bf2f((unsigned short)(xin[j] & 0xffffu));
            const float x1 = bf2f((unsigned short)(xin[j] >> 16));
            v0 = x0 * __builtin_amdgcn_rcpf(1.f + __expf(-v0));
            v1 = x1 * __builtin_amdgcn_rcpf(1.f + __expf(-v1));
            sm[m][2 * j] += v0; sm[m][2 * j + 1] += v1;
          }
          o[j] = pack_bf16(v0, v1);
        }
        *(v4u*)p = o;
      }
    }
  }

  if (MODE == 2) {
    // reduce over the 16 lanes that share each channel, then one atomic per channel
#pragma unroll
    for (int m = 0; m < 3; ++m)
#pragma unroll
      for (int v = 0; v < 8; ++v) {
        float s = sm[m][v];
        s += __shfl_xor(s, 1, 32);
        s += __shfl_xor(s, 2, 32);
        s += __shfl_xor(s, 4, 32);
        s += __shfl_xor(s, 8, 32);
        if (lm == 0) {
          const int co = 16 * m + 8 * hi + v;
          if (co < C_) atomicAdd(&partialRow[co], s);
        }
      }
  }
}

__global__ __launch_bounds__(THREADS) void timnet_main(
    const float* __restrict__ in,
    const float* __restrict__ c1w, const float* __restrict__ c1b,
    const float* __restrict__ fw_w, const float* __restrict__ fw_b,
    const float* __restrict__ fw_g, const float* __restrict__ fw_be,
    const float* __restrict__ fw_m, const float* __restrict__ fw_v,
    const float* __restrict__ bw_w, const float* __restrict__ bw_b,
    const float* __restrict__ bw_g, const float* __restrict__ bw_be,
    const float* __restrict__ bw_m, const float* __restrict__ bw_v,
    float* __restrict__ partial) {
  extern __shared__ unsigned short smem[];
  unsigned short* X = smem;            // [T_][CP] bf16 activations
  unsigned short* Y = smem + T_ * CP;  // [T_][CP] bf16 intermediate

  const int b   = blockIdx.x >> 1;
  const int dir = blockIdx.x & 1;
  const int tid = threadIdx.x;

  // ---- load (+optional time-reverse) + bf16-convert; layout matches global
  for (int t = tid; t < T_; t += THREADS) {
    const int trow = dir ? (T_ - 1 - t) : t;
    const float* row = in + ((size_t)b * T_ + trow) * C_;
    __builtin_prefetch(row + THREADS * C_, 0, 1);
    unsigned short* xr = X + t * CP;
#pragma unroll 1
    for (int c = 0; c < C_; ++c) xr[c] = f2bf(row[c]);
    xr[C_] = 0;                        // keep pad column zero
  }
  __syncthreads();

  // ---- conv1x1 (in-place: column t depends only on column t)
  conv_stage<0>(X, X, c1w, c1b, nullptr, nullptr, nullptr, nullptr, 1, nullptr);
  __syncthreads();

  const float* Wp  = dir ? bw_w  : fw_w;
  const float* Bp  = dir ? bw_b  : fw_b;
  const float* Gp  = dir ? bw_g  : fw_g;
  const float* BEp = dir ? bw_be : fw_be;
  const float* Mp  = dir ? bw_m  : fw_m;
  const float* Vp  = dir ? bw_v  : fw_v;

  for (int i = 0; i < NB_; ++i) {
    const int d = 1 << i;
    int o = (i * 2 + 0) * C_;
    conv_stage<1>(X, Y, Wp + (size_t)(i * 2 + 0) * C_ * C_ * 2,
                  Bp + o, Gp + o, BEp + o, Mp + o, Vp + o, d, nullptr);
    __syncthreads();
    o = (i * 2 + 1) * C_;
    conv_stage<2>(Y, X, Wp + (size_t)(i * 2 + 1) * C_ * C_ * 2,
                  Bp + o, Gp + o, BEp + o, Mp + o, Vp + o, d,
                  partial + (size_t)(i * B_ + b) * PCH);
    __syncthreads();
  }
}

__global__ void zero_ws(float* p, int n) {
  const int i = blockIdx.x * blockDim.x + threadIdx.x;
  if (i < n) p[i] = 0.f;
}

__global__ void logits_kernel(const float* __restrict__ part,
                              const float* __restrict__ wl,
                              const float* __restrict__ cw,
                              const float* __restrict__ cb,
                              float* __restrict__ out) {
  const int b = blockIdx.x * blockDim.x + threadIdx.x;
  if (b >= B_) return;
  float x2[C_];
  for (int c = 0; c < C_; ++c) {
    float s = 0.f;
    for (int i = 0; i < NB_; ++i) s += part[(size_t)(i * B_ + b) * PCH + c] * wl[i];
    x2[c] = s * (1.0f / (float)T_);
  }
  float lg[NC_];
  float mx = -1e30f;
  for (int o = 0; o < NC_; ++o) {
    float s = cb[o];
    for (int c = 0; c < C_; ++c) s += x2[c] * cw[o * C_ + c];
    lg[o] = s;
    mx = fmaxf(mx, s);
  }
  float se = 0.f;
  for (int o = 0; o < NC_; ++o) se += expf(lg[o] - mx);
  const float lse = mx + logf(se);
  for (int o = 0; o < NC_; ++o) out[b * NC_ + o] = lg[o] - lse;
}

extern "C" void kernel_launch(void* const* d_in, const int* in_sizes, int n_in,
                              void* d_out, int out_size, void* d_ws, size_t ws_size,
                              hipStream_t stream) {
  (void)in_sizes; (void)n_in; (void)out_size; (void)ws_size;
  const float* in    = (const float*)d_in[0];
  const float* c1w   = (const float*)d_in[1];
  const float* c1b   = (const float*)d_in[2];
  const float* fw_w  = (const float*)d_in[3];
  const float* fw_b  = (const float*)d_in[4];
  const float* fw_g  = (const float*)d_in[5];
  const float* fw_be = (const float*)d_in[6];
  const float* fw_m  = (const float*)d_in[7];
  const float* fw_v  = (const float*)d_in[8];
  const float* bw_w  = (const float*)d_in[9];
  const float* bw_b  = (const float*)d_in[10];
  const float* bw_g  = (const float*)d_in[11];
  const float* bw_be = (const float*)d_in[12];
  const float* bw_m  = (const float*)d_in[13];
  const float* bw_v  = (const float*)d_in[14];
  const float* wl    = (const float*)d_in[15];
  const float* cw    = (const float*)d_in[16];
  const float* cb    = (const float*)d_in[17];

  float* partial = (float*)d_ws;                 // [NB_][B_][PCH]
  const int n = NB_ * B_ * PCH;
  zero_ws<<<(n + 255) / 256, 256, 0, stream>>>(partial, n);

  const size_t smem = (size_t)2 * T_ * CP * sizeof(unsigned short);  // 320 KB
  timnet_main<<<B_ * 2, THREADS, smem, stream>>>(
      in, c1w, c1b, fw_w, fw_b, fw_g, fw_be, fw_m, fw_v,
      bw_w, bw_b, bw_g, bw_be, bw_m, bw_v, partial);

  logits_kernel<<<(B_ + 63) / 64, 64, 0, stream>>>(partial, wl, cw, cb, (float*)d_out);
}
